// mha_33122787786973
// MI455X (gfx1250) — matrix operand order repaired and checked by reading, not compiled, not run on hardware
//
#include <hip/hip_runtime.h>

#define DEVI __device__ __forceinline__

typedef __attribute__((ext_vector_type(16))) _Float16 v16h;
typedef __attribute__((ext_vector_type(8)))  _Float16 v8h;
typedef __attribute__((ext_vector_type(8)))  float    v8f;

static constexpr int Bn = 4, Ln = 2048, Hn = 2048, NHn = 16, HDn = 128;
static constexpr int TOK   = Bn * Ln;   // 8192 tokens
static constexpr int N_QKV = 3 * Hn;    // 6144

DEVI v16h ld_frag(const _Float16* p0, const _Float16* p1) {
  v8h lo = *(const v8h*)p0;
  v8h hi = *(const v8h*)p1;
  v16h r;
#pragma unroll
  for (int i = 0; i < 8; ++i) { r[i] = lo[i]; r[8 + i] = hi[i]; }
  return r;
}

DEVI v8f wmma16(v16h a, v16h b, v8f c) {
  return __builtin_amdgcn_wmma_f32_16x16x32_f16(false, a, false, b, (short)0, c, false, false);
}

// ---------------------------------------------------------------- converts
__global__ void k_cvt_f16(const float* __restrict__ in, _Float16* __restrict__ out, int n) {
  int i = blockIdx.x * blockDim.x + threadIdx.x;
  if (i < n) out[i] = (_Float16)in[i];
}

// Wt[n][k] = (f16) W[k][n]
__global__ void k_transpose_f16(const float* __restrict__ W, _Float16* __restrict__ Wt,
                                int K, int N) {
  long long i = (long long)blockIdx.x * blockDim.x + threadIdx.x;
  if (i >= (long long)K * N) return;
  int k = (int)(i % K);
  long long n = i / K;
  Wt[i] = (_Float16)W[(long long)k * N + n];
}

// ---------------------------------------------------------------- wave GEMM core
// One wave computes a 32x128 tile (2 A-frags x 8 B-frags = 16 accumulators).
// Ping-pong double buffering: two disjoint register sets, no staging copies,
// loads for step k+1 in flight while the 16 WMMAs for step k execute.
DEVI void ld_a2(const _Float16* pa0, const _Float16* pa1, int k0, v16h a[2]) {
  a[0] = ld_frag(pa0 + k0, pa0 + k0 + 16);
  a[1] = ld_frag(pa1 + k0, pa1 + k0 + 16);
}
DEVI void ld_b8(const _Float16* pb0, int ldb, int k0, v16h b[8]) {
#pragma unroll
  for (int t = 0; t < 8; ++t) {
    const _Float16* pb = pb0 + (size_t)t * 16 * ldb + k0;
    b[t] = ld_frag(pb, pb + 16);
  }
}

DEVI void wave_gemm_32x128(const _Float16* __restrict__ A, int lda,
                           const _Float16* __restrict__ Bt, int ldb,
                           int ksteps, int lane, v8f acc[2][8]) {
  const int m = lane & 15;
  const int h = lane >> 4;
  const _Float16* pa0 = A + (size_t)m * lda + h * 8;
  const _Float16* pa1 = A + (size_t)(m + 16) * lda + h * 8;
  const _Float16* pb0 = Bt + (size_t)m * ldb + h * 8;

  v16h a0[2], b0[8], a1[2], b1[8];
  ld_a2(pa0, pa1, 0, a0);
  ld_b8(pb0, ldb, 0, b0);

#pragma unroll 1
  for (int ks = 0; ks < ksteps; ks += 2) {
    __builtin_prefetch(pa0 + (ks + 2) * 32, 0, 3);
    ld_a2(pa0, pa1, (ks + 1) * 32, a1);
    ld_b8(pb0, ldb, (ks + 1) * 32, b1);
#pragma unroll
    for (int t = 0; t < 8; ++t) {
      acc[0][t] = wmma16(a0[0], b0[t], acc[0][t]);
      acc[1][t] = wmma16(a0[1], b0[t], acc[1][t]);
    }
    if (ks + 2 < ksteps) {
      ld_a2(pa0, pa1, (ks + 2) * 32, a0);
      ld_b8(pb0, ldb, (ks + 2) * 32, b0);
    }
#pragma unroll
    for (int t = 0; t < 8; ++t) {
      acc[0][t] = wmma16(a1[0], b1[t], acc[0][t]);
      acc[1][t] = wmma16(a1[1], b1[t], acc[1][t]);
    }
  }
}

// ---------------------------------------------------------------- QKV GEMM + scatter
// col n: part = n/2048, head = (n%2048)%16, d = (n%2048)/16.
// head = lane&15, d = (nblk&15)*8 + t -> Q/K pack into one v8h store per row.
__global__ void __launch_bounds__(256) k_gemm_qkv(
    const _Float16* __restrict__ Xh, const _Float16* __restrict__ WqkvT,
    const float* __restrict__ bqkv,
    _Float16* __restrict__ Qh, _Float16* __restrict__ Kh, _Float16* __restrict__ Vt) {
  const int lane = threadIdx.x & 31;
  const int wid  = blockIdx.x * (blockDim.x >> 5) + (threadIdx.x >> 5);
  const int NBLK = N_QKV / 128;          // 48
  const int nblk = wid % NBLK;
  const int mtile = wid / NBLK;          // 0..255 (32 rows each)
  v8f acc[2][8] = {};
  wave_gemm_32x128(Xh + (size_t)mtile * 32 * Hn, Hn,
                   WqkvT + (size_t)nblk * 128 * Hn, Hn, Hn / 32, lane, acc);
  const int m = lane & 15, h = lane >> 4;
  float biasv[8];
#pragma unroll
  for (int t = 0; t < 8; ++t) biasv[t] = bqkv[nblk * 128 + t * 16 + m];

  const int part = nblk >> 4;            // wave-uniform
  const int head = m;
  const int d0   = (nblk & 15) * 8;
  if (part < 2) {
    _Float16* dst = (part == 0) ? Qh : Kh;
#pragma unroll
    for (int g = 0; g < 2; ++g)
#pragma unroll
      for (int i = 0; i < 8; ++i) {
        const int row = mtile * 32 + g * 16 + i + 8 * h;
        const int bb  = row >> 11, l = row & 2047;
        v8h pk;
#pragma unroll
        for (int t = 0; t < 8; ++t) pk[t] = (_Float16)(acc[g][t][i] + biasv[t]);
        *(v8h*)(dst + ((size_t)(bb * NHn + head) * Ln + l) * HDn + d0) = pk;
      }
  } else {
#pragma unroll
    for (int g = 0; g < 2; ++g)
#pragma unroll
      for (int t = 0; t < 8; ++t)
#pragma unroll
        for (int i = 0; i < 8; ++i) {
          const int row = mtile * 32 + g * 16 + i + 8 * h;
          const int bb  = row >> 11, l = row & 2047;
          Vt[((size_t)(bb * NHn + head) * HDn + d0 + t) * Ln + l] =
              (_Float16)(acc[g][t][i] + biasv[t]);
        }
  }
}

// ---------------------------------------------------------------- rotary (in-place)
__global__ void k_rope(_Float16* __restrict__ X,
                       const float* __restrict__ cosb, const float* __restrict__ sinb) {
  long long idx = (long long)blockIdx.x * blockDim.x + threadIdx.x;
  const long long total = (long long)Bn * NHn * Ln * 64;
  if (idx >= total) return;
  const int d = (int)(idx & 63);
  long long r = idx >> 6;                 // row = bh*L + l
  const int l = (int)(r & 2047);
  const int head = (int)((r >> 11) & 15);
  const float c = cosb[l * (Hn / 2) + d * 16 + head];
  const float s = sinb[l * (Hn / 2) + d * 16 + head];
  _Float16* row = X + r * HDn;
  const float u1 = (float)row[d];
  const float u2 = (float)row[d + 64];
  row[d]      = (_Float16)( u1 * c + u2 * s);
  row[d + 64] = (_Float16)(-u1 * s + u2 * c);
}

// ---------------------------------------------------------------- flash attention
__global__ void __launch_bounds__(128) k_attn(
    const _Float16* __restrict__ Qh, const _Float16* __restrict__ Kh,
    const _Float16* __restrict__ Vt, _Float16* __restrict__ Yh) {
  __shared__ _Float16 plds[4 * 16 * 32];          // 1KB per wave scratch
  __builtin_amdgcn_s_wait_tensorcnt(0);
  const int lane  = threadIdx.x & 31;
  const int wslot = threadIdx.x >> 5;
  const int wid   = blockIdx.x * 4 + wslot;
  const int qt = wid & 127;
  const int bh = wid >> 7;
  const int b = bh >> 4, head = bh & 15;
  const int q0 = qt * 16;
  const _Float16* Q = Qh + (size_t)bh * Ln * HDn;
  const _Float16* K = Kh + (size_t)bh * Ln * HDn;
  const _Float16* V = Vt + (size_t)bh * HDn * Ln;
  const int m = lane & 15, h = lane >> 4;

  v16h qf[4];
#pragma unroll
  for (int kk = 0; kk < 4; ++kk) {
    const _Float16* pa = Q + (size_t)(q0 + m) * HDn + kk * 32 + h * 16;
    qf[kk] = ld_frag(pa, pa + 8);
  }

  v8f O[8] = {};
  float mrow[8], srow[8];
#pragma unroll
  for (int i = 0; i < 8; ++i) { mrow[i] = -3.0e38f; srow[i] = 0.f; }
  const float scale = 0.08838834764831845f;       // 1/sqrt(128)
  _Float16* pl = plds + wslot * 512;

  for (int n0 = 0; n0 < q0 + 16; n0 += 32) {
    // ---- batch-load all K fragments, then 8 S WMMAs
    v16h kb[8];
#pragma unroll
    for (int kk = 0; kk < 4; ++kk) {
      const _Float16* pb0 = K + (size_t)(n0 + m) * HDn + kk * 32 + h * 16;
      kb[kk] = ld_frag(pb0, pb0 + 8);
      const _Float16* pb1 = K + (size_t)(n0 + 16 + m) * HDn + kk * 32 + h * 16;
      kb[4 + kk] = ld_frag(pb1, pb1 + 8);
    }
    v8f S0 = {}, S1 = {};
#pragma unroll
    for (int kk = 0; kk < 4; ++kk) {
      S0 = wmma16(qf[kk], kb[kk], S0);
      S1 = wmma16(qf[kk], kb[4 + kk], S1);
    }
    // ---- issue V fragment loads now so they hide under the softmax math
    v16h vb[8];
#pragma unroll
    for (int dt = 0; dt < 8; ++dt) {
      const _Float16* pb = V + (size_t)(dt * 16 + m) * Ln + n0 + h * 16;
      vb[dt] = ld_frag(pb, pb + 8);
    }
    // ---- scale + causal mask
    const int col0 = n0 + m, col1 = n0 + 16 + m;
    float rm[8];
#pragma unroll
    for (int i = 0; i < 8; ++i) {
      const int row = q0 + i + 8 * h;
      float a0 = S0[i] * scale; if (col0 > row) a0 = -3.0e38f;
      float a1 = S1[i] * scale; if (col1 > row) a1 = -3.0e38f;
      S0[i] = a0; S1[i] = a1;
      rm[i] = fmaxf(a0, a1);
    }
#pragma unroll
    for (int i = 0; i < 8; ++i) {
      rm[i] = fmaxf(rm[i], __shfl_xor(rm[i], 1, 32));
      rm[i] = fmaxf(rm[i], __shfl_xor(rm[i], 2, 32));
      rm[i] = fmaxf(rm[i], __shfl_xor(rm[i], 4, 32));
      rm[i] = fmaxf(rm[i], __shfl_xor(rm[i], 8, 32));
    }
    float alpha[8], rs[8];
#pragma unroll
    for (int i = 0; i < 8; ++i) {
      const float mn = fmaxf(mrow[i], rm[i]);
      alpha[i] = __expf(mrow[i] - mn);
      mrow[i] = mn;
      const float p0 = __expf(S0[i] - mn);
      const float p1 = __expf(S1[i] - mn);
      S0[i] = p0; S1[i] = p1;
      rs[i] = p0 + p1;
    }
#pragma unroll
    for (int i = 0; i < 8; ++i) {
      rs[i] += __shfl_xor(rs[i], 1, 32);
      rs[i] += __shfl_xor(rs[i], 2, 32);
      rs[i] += __shfl_xor(rs[i], 4, 32);
      rs[i] += __shfl_xor(rs[i], 8, 32);
      srow[i] = srow[i] * alpha[i] + rs[i];
    }
#pragma unroll
    for (int t = 0; t < 8; ++t)
#pragma unroll
      for (int i = 0; i < 8; ++i) O[t][i] *= alpha[i];

    // ---- P (C-layout fp32) -> LDS (16x32 f16 row-major) -> A fragment
#pragma unroll
    for (int i = 0; i < 8; ++i) {
      const int row = i + 8 * h;
      pl[row * 32 + m]      = (_Float16)S0[i];
      pl[row * 32 + 16 + m] = (_Float16)S1[i];
    }
    const _Float16* pa = pl + m * 32 + h * 16;
    v16h pf = ld_frag(pa, pa + 8);
#pragma unroll
    for (int dt = 0; dt < 8; ++dt)
      O[dt] = wmma16(pf, vb[dt], O[dt]);
  }

  float inv[8];
#pragma unroll
  for (int i = 0; i < 8; ++i) inv[i] = 1.0f / srow[i];
#pragma unroll
  for (int dt = 0; dt < 8; ++dt)
#pragma unroll
    for (int i = 0; i < 8; ++i) {
      const int row = q0 + i + 8 * h;
      const int col = head * HDn + dt * 16 + m;
      Yh[((size_t)(b * Ln + row)) * Hn + col] = (_Float16)(O[dt][i] * inv[i]);
    }
}

// ---------------------------------------------------------------- FC2 + SiLU
__global__ void __launch_bounds__(256) k_gemm_fc2(
    const _Float16* __restrict__ Yh, const _Float16* __restrict__ WfcT,
    const float* __restrict__ bfc2, float* __restrict__ out) {
  const int lane = threadIdx.x & 31;
  const int wid  = blockIdx.x * (blockDim.x >> 5) + (threadIdx.x >> 5);
  const int NBLK = Hn / 128;             // 16
  const int nblk = wid % NBLK, mtile = wid / NBLK;
  v8f acc[2][8] = {};
  wave_gemm_32x128(Yh + (size_t)mtile * 32 * Hn, Hn,
                   WfcT + (size_t)nblk * 128 * Hn, Hn, Hn / 32, lane, acc);
  const int m = lane & 15, h = lane >> 4;
#pragma unroll
  for (int t = 0; t < 8; ++t) {
    const int col = nblk * 128 + t * 16 + m;
    const float bias = bfc2[col];
#pragma unroll
    for (int g = 0; g < 2; ++g)
#pragma unroll
      for (int i = 0; i < 8; ++i) {
        const int row = mtile * 32 + g * 16 + i + 8 * h;
        const float v = acc[g][t][i] + bias;
        out[(size_t)row * Hn + col] = v / (1.0f + __expf(-v));   // silu
      }
  }
}

// ---------------------------------------------------------------- launcher
extern "C" void kernel_launch(void* const* d_in, const int* in_sizes, int n_in,
                              void* d_out, int out_size, void* d_ws, size_t ws_size,
                              hipStream_t stream) {
  const float* x    = (const float*)d_in[0];
  const float* Wqkv = (const float*)d_in[1];
  const float* bqkv = (const float*)d_in[2];
  const float* Wfc2 = (const float*)d_in[3];
  const float* bfc2 = (const float*)d_in[4];
  const float* cosb = (const float*)d_in[5];
  const float* sinb = (const float*)d_in[6];

  char* ws = (char*)d_ws;
  _Float16* Xh    = (_Float16*)ws; ws += (size_t)TOK * Hn * 2;
  _Float16* WqkvT = (_Float16*)ws; ws += (size_t)N_QKV * Hn * 2;
  _Float16* WfcT  = (_Float16*)ws; ws += (size_t)Hn * Hn * 2;
  _Float16* Qh    = (_Float16*)ws; ws += (size_t)TOK * Hn * 2;
  _Float16* Kh    = (_Float16*)ws; ws += (size_t)TOK * Hn * 2;
  _Float16* Vt    = (_Float16*)ws; ws += (size_t)TOK * Hn * 2;
  _Float16* Yh    = (_Float16*)ws; ws += (size_t)TOK * Hn * 2;

  { int n = TOK * Hn;
    k_cvt_f16<<<(n + 255) / 256, 256, 0, stream>>>(x, Xh, n); }
  { long long n = (long long)Hn * N_QKV;
    k_transpose_f16<<<(unsigned)((n + 255) / 256), 256, 0, stream>>>(Wqkv, WqkvT, Hn, N_QKV); }
  { long long n = (long long)Hn * Hn;
    k_transpose_f16<<<(unsigned)((n + 255) / 256), 256, 0, stream>>>(Wfc2, WfcT, Hn, Hn); }
  { int waves = (TOK / 32) * (N_QKV / 128);          // 12288 waves
    k_gemm_qkv<<<waves / 8, 256, 0, stream>>>(Xh, WqkvT, bqkv, Qh, Kh, Vt); }
  { long long n = (long long)Bn * NHn * Ln * 64;
    k_rope<<<(unsigned)((n + 255) / 256), 256, 0, stream>>>(Qh, cosb, sinb);
    k_rope<<<(unsigned)((n + 255) / 256), 256, 0, stream>>>(Kh, cosb, sinb); }
  { int nwaves = Bn * NHn * (Ln / 16);               // 8192 waves
    k_attn<<<nwaves / 4, 128, 0, stream>>>(Qh, Kh, Vt, Yh); }
  { int waves = (TOK / 32) * (Hn / 128);             // 4096 waves
    k_gemm_fc2<<<waves / 8, 256, 0, stream>>>(Yh, WfcT, bfc2, (float*)d_out); }
}
